// LMheadModel_35424890257434
// MI455X (gfx1250) — matrix operand order repaired and checked by reading, not compiled, not run on hardware
//
#include <hip/hip_runtime.h>
#include <hip/hip_bf16.h>
#include <math.h>

// ---------------------------------------------------------------------------
// LM head (y = x @ W^T + b) + shifted cross-entropy loss, for MI455X (gfx1250)
//  1) convert W fp32 -> bf16 (also zero loss slot)
//  2) convert x fp32 [M][K] -> bf16 [M][K] (feeds WMMA B-layout contiguously)
//  3) WMMA bf16 GEMM (v_wmma_f32_16x16x32_bf16, fp32 accum), software-pipelined
//     (double-buffered fragments + W-stream prefetch); bias fused at the
//     coalesced store; epilogue emits per-(row, Nblock) partial softmax so the
//     loss pass never re-reads the 823 MB logits tensor
//  4) loss: logsumexp-merge 393 partials per shifted row + label logit
// ---------------------------------------------------------------------------

typedef __bf16 bf16_t;
typedef __attribute__((ext_vector_type(16))) __bf16 v16bf;
typedef __attribute__((ext_vector_type(8)))  __bf16 v8bf;
typedef __attribute__((ext_vector_type(8)))  float  v8f;

static constexpr int B = 2;
static constexpr int S = 2048;
static constexpr int K = 1024;       // E
static constexpr int V = 50257;
static constexpr int M = B * S;      // 4096 sequence rows

static constexpr int MT = 64;        // seq rows per workgroup tile
static constexpr int NT = 128;       // vocab cols per workgroup tile
static constexpr int NBLK = (V + NT - 1) / NT;   // 393 N-blocks
static constexpr int LDS_STRIDE = MT + 4;        // 68: conflict-free staging

// ---------------------------------------------------------------------------
// Kernel 1: W fp32 -> bf16 (row-major [V][K] preserved), zero the loss slot.
// ---------------------------------------------------------------------------
__global__ __launch_bounds__(256) void convert_w_kernel(
    const float* __restrict__ W, bf16_t* __restrict__ Wb,
    float* __restrict__ lossSlot) {
  if (blockIdx.x == 0 && threadIdx.x == 0) *lossSlot = 0.0f;
  const size_t n4 = (size_t)V * K / 4;
  for (size_t i = (size_t)blockIdx.x * blockDim.x + threadIdx.x; i < n4;
       i += (size_t)gridDim.x * blockDim.x) {
    const float4 v = ((const float4*)W)[i];
    union { bf16_t h[4]; uint2 u; } r;
    r.h[0] = (bf16_t)v.x; r.h[1] = (bf16_t)v.y;
    r.h[2] = (bf16_t)v.z; r.h[3] = (bf16_t)v.w;
    *(uint2*)(Wb + 4 * i) = r.u;
  }
}

// ---------------------------------------------------------------------------
// Kernel 2: x fp32 [M][K] -> bf16 [M][K], same layout, via LDS 64x64 tiles.
// ---------------------------------------------------------------------------
__global__ __launch_bounds__(256) void transpose_x_kernel(
    const float* __restrict__ x, bf16_t* __restrict__ xT) {
  __shared__ bf16_t t[64][66];          // stride 66 elems -> conflict-free
  const int m0 = blockIdx.x * 64;
  const int k0 = blockIdx.y * 64;
  const int tr = threadIdx.x >> 6;      // 0..3
  const int tc = threadIdx.x & 63;      // 0..63
#pragma unroll
  for (int r = 0; r < 16; ++r) {
    const int ml = tr + r * 4;
    t[ml][tc] = (bf16_t)x[(size_t)(m0 + ml) * K + k0 + tc];
  }
  __syncthreads();
#pragma unroll
  for (int r = 0; r < 16; ++r) {
    const int kl = tr + r * 4;
    xT[(size_t)(m0 + kl) * K + k0 + tc] = t[kl][tc];
  }
}

// ---------------------------------------------------------------------------
// WMMA fragment helpers.
// A (16-bit, 16x32): lane<16 holds row M=lane with K {c..c+7}, c in {0,16};
// lane>=16 holds K {8..15, 24..31} -> two 16B loads + register concat.
// ---------------------------------------------------------------------------
__device__ __forceinline__ v16bf load_a_frag(const bf16_t* p) {
  const v8bf lo = *(const v8bf*)p;
  const v8bf hi = *(const v8bf*)(p + 16);
  return __builtin_shufflevector(lo, hi, 0, 1, 2, 3, 4, 5, 6, 7,
                                 8, 9, 10, 11, 12, 13, 14, 15);
}

__device__ __forceinline__ void wmma4(const v16bf a[2], const v16bf b[2],
                                      v8f acc[2][2]) {
#pragma unroll
  for (int vi = 0; vi < 2; ++vi)
#pragma unroll
    for (int mi = 0; mi < 2; ++mi)
      acc[vi][mi] = __builtin_amdgcn_wmma_f32_16x16x32_bf16(
          false, a[vi], false, b[mi], (short)0, acc[vi][mi], false, false);
}

// ---------------------------------------------------------------------------
// Kernel 3: WMMA GEMM + fused bias + per-(row, Nblock) partial softmax.
// D(vocab x seq) = A(W tile 16x32) x B(xT tile 32x16).
// B layout: lane supplies n = lane&15 with k in the same order as A
// (xT stored [M][K]) -> load_a_frag.
// ---------------------------------------------------------------------------
__global__ __launch_bounds__(256) void gemm_kernel(
    const bf16_t* __restrict__ xT,    // [M][K] bf16
    const bf16_t* __restrict__ Wb,    // [V][K] bf16
    const float* __restrict__ bias,   // [V]
    float* __restrict__ y,            // [M][V]
    float2* __restrict__ partials) {  // [M][NBLK] (max, sumexp)
  __shared__ float tile[NT * LDS_STRIDE];

  const int tid  = threadIdx.x;
  const int lane = tid & 31;
  const int wave = tid >> 5;       // 0..7
  const int wm   = wave & 1;       // seq strip (2 x 32)
  const int wv   = wave >> 1;      // vocab strip (4 x 32)
  const int m0   = blockIdx.x * MT;    // seq base (fast dim -> W-strip L2 reuse)
  const int v0   = blockIdx.y * NT;    // vocab base

  const int laneLo = lane & 15;
  const int laneHi = lane >> 4;    // 0 or 1
  const int seqBase = m0 + wm * 32;
  const int vocBase = v0 + wv * 32;

  const int kOffA = laneHi * 8;              // A chunk start within 32-K step
  const int kIdxB = laneLo + laneHi * 16;    // unused; B loads use kOffA like A

  int vr0 = vocBase + laneLo;        if (vr0 >= V) vr0 = V - 1;
  int vr1 = vocBase + 16 + laneLo;   if (vr1 >= V) vr1 = V - 1;

  const bf16_t* pA0 = Wb + (size_t)vr0 * K + kOffA;
  const bf16_t* pA1 = Wb + (size_t)vr1 * K + kOffA;
  const bf16_t* pB0 = xT + (size_t)(seqBase + laneLo) * K + kOffA;
  const bf16_t* pB1 = pB0 + (size_t)16 * K;

  v8f acc[2][2] = {};
  v16bf aA[2], bA[2], aB[2], bB[2];

  // prologue: fragments for k = 0
  aA[0] = load_a_frag(pA0);
  aA[1] = load_a_frag(pA1);
  bA[0] = load_a_frag(pB0);
  bA[1] = load_a_frag(pB1);

  for (int k0 = 0; k0 < K; k0 += 64) {
    // prefetch W stream 512B ahead (global_prefetch_b8)
    if (k0 + 256 < K) {
      __builtin_prefetch(pA0 + 256, 0, 0);
      __builtin_prefetch(pA1 + 256, 0, 0);
    }
    // issue loads for k0+32 before consuming k0 fragments
    aB[0] = load_a_frag(pA0 + 32);
    aB[1] = load_a_frag(pA1 + 32);
    bB[0] = load_a_frag(pB0 + 32);
    bB[1] = load_a_frag(pB1 + 32);
    wmma4(aA, bA, acc);

    pA0 += 64; pA1 += 64;
    pB0 += 64; pB1 += 64;
    if (k0 + 64 < K) {             // issue loads for k0+64 before k0+32 math
      aA[0] = load_a_frag(pA0);
      aA[1] = load_a_frag(pA1);
      bA[0] = load_a_frag(pB0);
      bA[1] = load_a_frag(pB1);
    }
    wmma4(aB, bB, acc);
  }

  // Stage accumulators in LDS: tile[vocab_local][seq_local]
#pragma unroll
  for (int vi = 0; vi < 2; ++vi)
#pragma unroll
    for (int mi = 0; mi < 2; ++mi)
#pragma unroll
      for (int j = 0; j < 8; ++j) {
        const int vl = wv * 32 + vi * 16 + j + laneHi * 8;   // D row = vocab
        const int sl = wm * 32 + mi * 16 + laneLo;           // D col = seq
        tile[vl * LDS_STRIDE + sl] = acc[vi][mi][j];
      }
  __syncthreads();

  // Writeback: each wave owns one 128-col row slice per iteration (lane u
  // covers cols 4u..4u+3) -> coalesced float4 stores with fused bias, plus a
  // wave-level (max, sum-exp) partial softmax for the loss pass.
  const int vq = lane * 4;             // 0,4,...,124
  for (int it = 0; it < MT / 8; ++it) {
    const int sl = it * 8 + wave;
    float4 val;
    val.x = tile[(vq + 0) * LDS_STRIDE + sl];
    val.y = tile[(vq + 1) * LDS_STRIDE + sl];
    val.z = tile[(vq + 2) * LDS_STRIDE + sl];
    val.w = tile[(vq + 3) * LDS_STRIDE + sl];
    const int v = v0 + vq;
    float* dst = y + (size_t)(m0 + sl) * V + v;
    float w0, w1, w2, w3;
    if (v + 3 < V) {
      const float4 bb = *(const float4*)(bias + v);
      w0 = val.x + bb.x; w1 = val.y + bb.y;
      w2 = val.z + bb.z; w3 = val.w + bb.w;
      const float4 st4 = {w0, w1, w2, w3};
      *(float4*)dst = st4;
    } else {
      const float vv[4] = {val.x, val.y, val.z, val.w};
      float wm4[4];
      for (int c = 0; c < 4; ++c) {
        if (v + c < V) {
          wm4[c] = vv[c] + bias[v + c];
          dst[c] = wm4[c];
        } else {
          wm4[c] = -__builtin_inff();       // masked: exp(-inf)=0
        }
      }
      w0 = wm4[0]; w1 = wm4[1]; w2 = wm4[2]; w3 = wm4[3];
    }
    // wave-level max
    float wmax = fmaxf(fmaxf(w0, w1), fmaxf(w2, w3));
#pragma unroll
    for (int off = 16; off > 0; off >>= 1)
      wmax = fmaxf(wmax, __shfl_xor(wmax, off, 32));
    // wave-level sum of exp(x - max)
    float ssum = __expf(w0 - wmax) + __expf(w1 - wmax) +
                 __expf(w2 - wmax) + __expf(w3 - wmax);
#pragma unroll
    for (int off = 16; off > 0; off >>= 1)
      ssum += __shfl_xor(ssum, off, 32);
    if (lane == 0)
      partials[(size_t)(m0 + sl) * NBLK + blockIdx.y] = make_float2(wmax, ssum);
  }
}

// ---------------------------------------------------------------------------
// Kernel 4: shifted cross-entropy from partials. One block per row r in
// [0, B*(S-1)); merges NBLK (max, sumexp) pairs with streaming logsumexp.
// ---------------------------------------------------------------------------
__global__ __launch_bounds__(128) void loss_kernel(
    const float* __restrict__ y, const float2* __restrict__ partials,
    const int* __restrict__ labels, float* __restrict__ lossSlot) {
  __shared__ float rm[128], rs[128];
  const int r  = blockIdx.x;                // 0 .. B*(S-1)-1
  const int bb = r / (S - 1);
  const int s  = r % (S - 1);
  const size_t mrow = (size_t)bb * S + s;

  float mx = -__builtin_inff();
  float sm = 0.0f;
  for (int j = threadIdx.x; j < NBLK; j += 128) {
    const float2 p = partials[mrow * NBLK + j];
    const float nm = fmaxf(mx, p.x);
    sm = sm * __expf(mx - nm) + p.y * __expf(p.x - nm);
    mx = nm;
  }
  rm[threadIdx.x] = mx;
  rs[threadIdx.x] = sm;
  __syncthreads();
  for (int st = 64; st > 0; st >>= 1) {
    if (threadIdx.x < st) {
      const float m2 = rm[threadIdx.x + st];
      const float s2 = rs[threadIdx.x + st];
      const float nm = fmaxf(rm[threadIdx.x], m2);
      rs[threadIdx.x] =
          rs[threadIdx.x] * __expf(rm[threadIdx.x] - nm) + s2 * __expf(m2 - nm);
      rm[threadIdx.x] = nm;
    }
    __syncthreads();
  }
  if (threadIdx.x == 0) {
    const int label = labels[mrow + 1];     // b*S + (s+1), same batch row
    const float lab = y[mrow * (size_t)V + label];
    const float lse = rm[0] + logf(rs[0]);
    atomicAdd(lossSlot, (lse - lab) * (1.0f / (float)(B * (S - 1))));
  }
}

// ---------------------------------------------------------------------------
// Launch. Inputs (dict order): x [B,S,E] f32, _y [B,S] int, W [V,E] f32,
// b [V] f32. d_out = y (B*S*V floats) ++ loss (1 float).
// Workspace: xT bf16 (8 MB) | W bf16 (~98 MiB) | partials (12.9 MB).
// ---------------------------------------------------------------------------
extern "C" void kernel_launch(void* const* d_in, const int* in_sizes, int n_in,
                              void* d_out, int out_size, void* d_ws,
                              size_t ws_size, hipStream_t stream) {
  const float* x      = (const float*)d_in[0];
  const int*   labels = (const int*)d_in[1];
  const float* W      = (const float*)d_in[2];
  const float* bias   = (const float*)d_in[3];

  float* y        = (float*)d_out;
  float* lossSlot = y + (size_t)M * V;

  const size_t xT_bytes = (size_t)K * M * sizeof(bf16_t);   // 8 MB
  const size_t Wb_bytes = (size_t)V * K * sizeof(bf16_t);   // ~98 MiB
  bf16_t* xT = (bf16_t*)d_ws;
  bf16_t* Wb = (bf16_t*)((char*)d_ws + xT_bytes);
  float2* partials = (float2*)((char*)d_ws + xT_bytes + Wb_bytes);

  convert_w_kernel<<<4096, 256, 0, stream>>>(W, Wb, lossSlot);
  transpose_x_kernel<<<dim3(M / 64, K / 64), 256, 0, stream>>>(x, xT);
  gemm_kernel<<<dim3(M / MT, NBLK), 256, 0, stream>>>(xT, Wb, bias, y, partials);
  loss_kernel<<<B * (S - 1), 128, 0, stream>>>(y, partials, labels, lossSlot);
}
